// MoELayer_87832081203913
// MI455X (gfx1250) — matrix operand order repaired and checked by reading, not compiled, not run on hardware
//
#include <hip/hip_runtime.h>
#include <hip/hip_bf16.h>

// ---------------------------------------------------------------------------
// MoE (N=16384, D=1024, E=8, top-2) for gfx1250 (MI455X).
// Gather/dispatch formulation: only the 2*N selected (token, expert) pairs are
// computed (68.7 GFLOP instead of the reference's dense 275 GFLOP), using
// v_wmma_f32_16x16x32_bf16 for the expert GEMMs. expert_w (16 MB as bf16)
// stays hot in the 192 MB L2; x is cast to bf16 once in the router pass.
// ---------------------------------------------------------------------------

#define NTOK   16384
#define DMODEL 1024
#define NEXP   8
#define TOPK   2
#define MB     16              // gathered-token rows per GEMM block
#define AST    (DMODEL + 8)    // LDS row stride in bf16 elems (pad kills bank conflicts)

typedef __bf16 v16bf __attribute__((ext_vector_type(16)));
typedef __bf16 v8bf  __attribute__((ext_vector_type(8)));
typedef __bf16 v4bf  __attribute__((ext_vector_type(4)));
typedef float  v8f   __attribute__((ext_vector_type(8)));

__device__ __forceinline__ void atomAddF32(float* p, float v) {
  __hip_atomic_fetch_add(p, v, __ATOMIC_RELAXED, __HIP_MEMORY_SCOPE_AGENT);
}

// --- 0. small zero kernels ---------------------------------------------------
__global__ void zero_meta_kernel(int* counts) {
  if (threadIdx.x < NEXP) counts[threadIdx.x] = 0;
}

__global__ void zero_out_kernel(float4* out4, int n4) {
  const float4 z = make_float4(0.f, 0.f, 0.f, 0.f);
  for (int i = blockIdx.x * blockDim.x + threadIdx.x; i < n4;
       i += gridDim.x * blockDim.x)
    out4[i] = z;
}

// --- 1. f32 -> bf16 cast (expert weights) -----------------------------------
__global__ void cvt_bf16_kernel(const float4* __restrict__ src,
                                v4bf* __restrict__ dst, int n4) {
  for (int i = blockIdx.x * blockDim.x + threadIdx.x; i < n4;
       i += gridDim.x * blockDim.x) {
    float4 v = src[i];
    v4bf o;
    o[0] = (__bf16)v.x; o[1] = (__bf16)v.y; o[2] = (__bf16)v.z; o[3] = (__bf16)v.w;
    dst[i] = o;
  }
}

// --- 2. router: logits, top-2 softmax gates, histogram, x -> bf16 ------------
__global__ __launch_bounds__(256) void router_kernel(
    const float* __restrict__ x, const float* __restrict__ rw,
    const float* __restrict__ rb, __bf16* __restrict__ xb,
    int* __restrict__ topi, float* __restrict__ topg, int* __restrict__ counts) {
  __shared__ float rws[NEXP * DMODEL];  // 32 KB
  const int t = threadIdx.x;
  for (int i = t * 4; i < NEXP * DMODEL; i += 256 * 4)
    *(float4*)(rws + i) = *(const float4*)(rw + i);
  __syncthreads();

  const int lane = t & 31, wv = t >> 5;
  const int n = blockIdx.x * 8 + wv;   // one wave per token
  const float* xr = x + (size_t)n * DMODEL;

  float acc[NEXP];
#pragma unroll
  for (int e = 0; e < NEXP; ++e) acc[e] = 0.f;

  for (int d = lane * 4; d < DMODEL; d += 128) {
    float4 xv = *(const float4*)(xr + d);
    v4bf xc;
    xc[0] = (__bf16)xv.x; xc[1] = (__bf16)xv.y;
    xc[2] = (__bf16)xv.z; xc[3] = (__bf16)xv.w;
    *(v4bf*)(xb + (size_t)n * DMODEL + d) = xc;   // fused bf16 cast of x
#pragma unroll
    for (int e = 0; e < NEXP; ++e) {
      float4 w = *(const float4*)(rws + e * DMODEL + d);
      acc[e] += xv.x * w.x + xv.y * w.y + xv.z * w.z + xv.w * w.w;
    }
  }
  // wave32 butterfly reduce (all lanes end with full sums)
#pragma unroll
  for (int off = 16; off >= 1; off >>= 1)
#pragma unroll
    for (int e = 0; e < NEXP; ++e) acc[e] += __shfl_xor(acc[e], off, 32);

  float lg[NEXP];
#pragma unroll
  for (int e = 0; e < NEXP; ++e) lg[e] = acc[e] + rb[e];

  int i0 = 0;
#pragma unroll
  for (int e = 1; e < NEXP; ++e) if (lg[e] > lg[i0]) i0 = e;  // ties -> low idx
  int i1 = (i0 == 0) ? 1 : 0;
#pragma unroll
  for (int e = 0; e < NEXP; ++e) if (e != i0 && lg[e] > lg[i1]) i1 = e;

  // renormalized top-2 softmax == softmax over the two top logits
  float p1 = __expf(lg[i1] - lg[i0]);
  float s  = 1.f + p1;
  if (lane == 0) {
    topi[2 * n] = i0;  topi[2 * n + 1] = i1;
    topg[2 * n] = 1.f / s;  topg[2 * n + 1] = p1 / s;
    atomicAdd(&counts[i0], 1);
    atomicAdd(&counts[i1], 1);
  }
}

// --- 3. prefix sum over 8 expert counts --------------------------------------
__global__ void offsets_kernel(const int* __restrict__ counts,
                               int* __restrict__ offs, int* __restrict__ cursor) {
  if (threadIdx.x == 0 && blockIdx.x == 0) {
    int s = 0;
    for (int e = 0; e < NEXP; ++e) { offs[e] = s; cursor[e] = s; s += counts[e]; }
    offs[NEXP] = s;   // == 2*N
  }
}

// --- 4. scatter tokens into expert-contiguous slots --------------------------
__global__ void scatter_kernel(const int* __restrict__ topi,
                               const float* __restrict__ topg,
                               int* __restrict__ cursor, int* __restrict__ tok,
                               float* __restrict__ gat) {
  int n = blockIdx.x * blockDim.x + threadIdx.x;
  if (n >= NTOK) return;
#pragma unroll
  for (int k = 0; k < TOPK; ++k) {
    int e = topi[2 * n + k];
    int pos = atomicAdd(&cursor[e], 1);
    tok[pos] = n;
    gat[pos] = topg[2 * n + k];
  }
}

// --- 5. per-expert gathered GEMM with bf16 WMMA ------------------------------
// Block: 256 threads = 8 waves. Grid: (row-tiles, expert).
// Block computes 16 gathered rows x all 1024 output cols (8 col passes of
// 8 waves * 16 cols). A tile (16 x 1024 bf16) staged once in LDS.
__global__ __launch_bounds__(256) void moe_gemm_kernel(
    const __bf16* __restrict__ xb, const __bf16* __restrict__ wb,
    const float* __restrict__ eb, const int* __restrict__ offs,
    const int* __restrict__ tok, const float* __restrict__ gat,
    float* __restrict__ out) {
  const int e = blockIdx.y;
  const int seg0 = offs[e], seg1 = offs[e + 1];
  const int tile0 = seg0 + blockIdx.x * MB;
  if (tile0 >= seg1) return;

  __shared__ __bf16 Alds[MB * AST];   // 33 KB
  __shared__ int   tokS[MB];
  __shared__ float gatS[MB];

  const int t = threadIdx.x;
  if (t < MB) {
    int gr = tile0 + t;
    tokS[t] = (gr < seg1) ? tok[gr] : 0;    // pad rows -> token 0, gate 0
    gatS[t] = (gr < seg1) ? gat[gr] : 0.f;
  }
  __syncthreads();

  {  // cooperative gathered stage of A (16 rows x 1024 bf16)
    int row = t >> 4;
    int cb  = (t & 15) * 64;
    const v8bf* src = (const v8bf*)(xb + (size_t)tokS[row] * DMODEL + cb);
    v8bf* dst = (v8bf*)(Alds + row * AST + cb);
#pragma unroll
    for (int i = 0; i < 8; ++i) dst[i] = src[i];
  }
  __syncthreads();

  const int lane = t & 31, wv = t >> 5;
  const int half = lane >> 4;          // which K half this lane holds
  const int lr   = lane & 15;

  for (int cp = 0; cp < 8; ++cp) {
    const int col = cp * 128 + wv * 16 + lr;            // this lane's N index
    const __bf16* wrow =
        wb + ((size_t)e * DMODEL + col) * DMODEL + half * 16;  // W_e[col][*]
    v8f c = {};

    for (int kb = 0; kb < DMODEL; kb += 32) {
      // B fragment (32x16): lane<16 holds K=kb..kb+15, lane>=16 K=kb+16..kb+31
      v8bf b0 = *(const v8bf*)(wrow + kb);
      v8bf b1 = *(const v8bf*)(wrow + kb + 8);
      // A fragment (16x32): lane<16 holds K=kb..kb+15, lane>=16 K=kb+16..kb+31
      const __bf16* arow = Alds + lr * AST + kb + half * 16;
      v8bf a0 = *(const v8bf*)(arow);
      v8bf a1 = *(const v8bf*)(arow + 8);
      v16bf A, B;
#pragma unroll
      for (int i = 0; i < 8; ++i) {
        A[i] = a0[i]; A[i + 8] = a1[i];
        B[i] = b0[i]; B[i + 8] = b1[i];
      }
      c = __builtin_amdgcn_wmma_f32_16x16x32_bf16(
          false, A, false, B, (short)0, c, false, false);
    }

    const float bias = eb[e * DMODEL + col];
#pragma unroll
    for (int r = 0; r < 8; ++r) {
      int m  = r + 8 * half;            // C layout: VGPR r -> row r (+8 hi half)
      int gr = tile0 + m;
      if (gr < seg1) {
        float v = gatS[m] * (c[r] + bias);
        atomAddF32(out + (size_t)tokS[m] * DMODEL + col, v);  // exactly 2 adds/elem
      }
    }
  }
}

// ---------------------------------------------------------------------------
extern "C" void kernel_launch(void* const* d_in, const int* in_sizes, int n_in,
                              void* d_out, int out_size, void* d_ws, size_t ws_size,
                              hipStream_t stream) {
  const float* x   = (const float*)d_in[0];   // [N, D]
  const float* rw  = (const float*)d_in[1];   // [E, D]
  const float* rb  = (const float*)d_in[2];   // [E]
  const float* ew  = (const float*)d_in[3];   // [E, D, D]
  const float* ebv = (const float*)d_in[4];   // [E, D]
  float* out = (float*)d_out;                 // [N, D]

  // workspace carve-up (≈48.5 MB total)
  char* p = (char*)d_ws;
  auto take = [&](size_t bytes) { char* r = p; p += (bytes + 255) & ~size_t(255); return r; };
  __bf16* xb    = (__bf16*)take((size_t)NTOK * DMODEL * 2);        // 32 MB
  __bf16* wbuf  = (__bf16*)take((size_t)NEXP * DMODEL * DMODEL * 2); // 16 MB
  int*    topi  = (int*)   take((size_t)NTOK * TOPK * 4);
  float*  topg  = (float*) take((size_t)NTOK * TOPK * 4);
  int*    tok   = (int*)   take((size_t)NTOK * TOPK * 4);
  float*  gat   = (float*) take((size_t)NTOK * TOPK * 4);
  int*    cnts  = (int*)   take(NEXP * 4);
  int*    offs  = (int*)   take((NEXP + 1) * 4);
  int*    curs  = (int*)   take(NEXP * 4);

  zero_meta_kernel<<<1, 64, 0, stream>>>(cnts);

  const int wn4 = NEXP * DMODEL * DMODEL / 4;
  cvt_bf16_kernel<<<2048, 256, 0, stream>>>((const float4*)ew, (v4bf*)wbuf, wn4);

  router_kernel<<<NTOK / 8, 256, 0, stream>>>(x, rw, rb, xb, topi, topg, cnts);

  offsets_kernel<<<1, 1, 0, stream>>>(cnts, offs, curs);

  scatter_kernel<<<NTOK / 256, 256, 0, stream>>>(topi, topg, curs, tok, gat);

  const int on4 = NTOK * DMODEL / 4;
  zero_out_kernel<<<4096, 256, 0, stream>>>((float4*)out, on4);

  dim3 grid(NTOK / MB, NEXP);   // over-provisioned; blocks past segment end exit
  moe_gemm_kernel<<<grid, 256, 0, stream>>>(xb, wbuf, ebv, offs, tok, gat, out);
}
